// MultiwayAttention_61125974557707
// MI455X (gfx1250) — matrix operand order repaired and checked by reading, not compiled, not run on hardware
//
#include <hip/hip_runtime.h>

// ---------------------------------------------------------------------------
// MI455X (gfx1250) multiway attention forward.
// All GEMM work goes through v_wmma_f32_16x16x32_bf16 (wave32, 16x16x32).
// fp32 inputs are converted once to bf16; accumulation stays fp32.
// GEMMs are register-blocked (4x2 tiles/wave) to push the inner loop from
// VMEM-issue-bound toward matrix-core-bound.
// ---------------------------------------------------------------------------

typedef __bf16 bf16;
typedef __attribute__((ext_vector_type(16))) __bf16 v16bf;
typedef __attribute__((ext_vector_type(8)))  __bf16 v8bf;
typedef __attribute__((ext_vector_type(8)))  float  v8f;

static constexpr int B_   = 2;
static constexpr int S_   = 2048;
static constexpr int D_   = 4096;
static constexpr int H_   = 32;
static constexpr int KV_  = 8;
static constexpr int HD_  = 128;
static constexpr int M_   = B_ * S_;     // 4096 token rows
static constexpr int NQ_  = H_ * HD_;    // 4096
static constexpr int NKV_ = KV_ * HD_;   // 1024

// ---------------- scalar bf16 helpers (bit-level, round-to-nearest-even) ----
__device__ inline bf16 f2bf(float f) {
  unsigned u = __builtin_bit_cast(unsigned, f);
  unsigned r = u + 0x7FFFu + ((u >> 16) & 1u);
  unsigned short h = (unsigned short)(r >> 16);
  return __builtin_bit_cast(bf16, h);
}
__device__ inline float bf2f(bf16 b) {
  unsigned short h = __builtin_bit_cast(unsigned short, b);
  return __builtin_bit_cast(float, (unsigned)h << 16);
}

// ---------------- WMMA fragment loaders (wave32, contiguous k order shared by A and B) --------
// A (16x32 bf16): lanes 0-15 hold row M=lane, K = k0..k0+15;
//                 lanes 16-31 hold row M=lane-16, K = k0+16..k0+31.
__device__ inline v16bf load_a_frag(const bf16* base, int ldk, int row, int k0, int lane) {
  int hi = lane >> 4;
  long r = row + (lane & 15);
  return *(const v16bf*)(base + r * (long)ldk + k0 + hi * 16);
}

// B (32x16 bf16) from a K-major ("transposed", N x K row-major) matrix:
// lanes 0-15 hold column N=col+lane, K = k0..k0+15; lanes 16-31: K = k0+16..k0+31.
__device__ inline v16bf load_b_frag(const bf16* baseT, int ldk, int col, int k0, int lane) {
  int hi = lane >> 4;
  long n = col + (lane & 15);
  return *(const v16bf*)(baseT + n * (long)ldk + k0 + hi * 16);
}

__device__ inline v8f wmma_bf16(v16bf a, v16bf b, v8f c) {
  return __builtin_amdgcn_wmma_f32_16x16x32_bf16(false, a, false, b, (short)0, c, false, false);
}

// ---------------- prep kernels ----------------------------------------------
__global__ void convert_f32_bf16_kernel(const float* __restrict__ in,
                                        bf16* __restrict__ out, int n) {
  int i = blockIdx.x * blockDim.x + threadIdx.x;
  if (i < n) out[i] = f2bf(in[i]);
}

// in: K x N row-major fp32  ->  out: N x K row-major bf16 (K-major weights)
__global__ void transpose_to_bf16_kernel(const float* __restrict__ in,
                                         bf16* __restrict__ out, int K, int N) {
  int i = blockIdx.x * blockDim.x + threadIdx.x;
  if (i < K * N) {
    int k = i / N, n = i % N;
    out[(long)n * K + k] = f2bf(in[i]);
  }
}

__global__ void rope_tables_kernel(const int* __restrict__ pos,
                                   float* __restrict__ cosT,
                                   float* __restrict__ sinT) {
  int i = blockIdx.x * blockDim.x + threadIdx.x;   // over S_*HD_
  if (i < S_ * HD_) {
    int s = i / HD_, d = i % HD_;
    float inv_freq = powf(10000.0f, -(float)(2 * (d % (HD_ / 2))) / (float)HD_);
    float ang = (float)pos[s] * inv_freq;
    cosT[i] = cosf(ang);
    sinT[i] = sinf(ang);
  }
}

// ---------------- register-blocked WMMA GEMM --------------------------------
// Wave computes (MT*16) x (NT*16). Waves within a block stride M so the
// block shares the streamed B (weight) fragments through L1/L2.
// Block tile: (4*MT*16) rows x (NT*16) cols.
template <int MT, int NT, bool F32OUT>
__global__ __launch_bounds__(128) void gemm_tiled_kernel(
    const bf16* __restrict__ A, const bf16* __restrict__ WT,
    void* __restrict__ Cv, int M, int N, int K) {
  int lane  = threadIdx.x & 31;
  int wave  = threadIdx.x >> 5;
  int tileN = blockIdx.x * (NT * 16);
  int tileM = (blockIdx.y * 4 + wave) * (MT * 16);

  v8f acc[MT][NT];
#pragma unroll
  for (int mi = 0; mi < MT; ++mi)
#pragma unroll
    for (int ni = 0; ni < NT; ++ni) acc[mi][ni] = (v8f){};

  for (int k0 = 0; k0 < K; k0 += 32) {
    __builtin_prefetch(A + (long)(tileM + (lane & 15)) * K + k0 + 256, 0, 1);
    __builtin_prefetch(WT + (long)(tileN + (lane & 15)) * K + k0 + 256, 0, 1);
    v16bf a[MT], b[NT];
#pragma unroll
    for (int mi = 0; mi < MT; ++mi)
      a[mi] = load_a_frag(A, K, tileM + mi * 16, k0, lane);
#pragma unroll
    for (int ni = 0; ni < NT; ++ni)
      b[ni] = load_b_frag(WT, K, tileN + ni * 16, k0, lane);
#pragma unroll
    for (int mi = 0; mi < MT; ++mi)
#pragma unroll
      for (int ni = 0; ni < NT; ++ni)
        acc[mi][ni] = wmma_bf16(a[mi], b[ni], acc[mi][ni]);
  }

  int hi = lane >> 4, cl = lane & 15;
#pragma unroll
  for (int mi = 0; mi < MT; ++mi)
#pragma unroll
    for (int ni = 0; ni < NT; ++ni)
#pragma unroll
      for (int r = 0; r < 8; ++r) {
        long row = tileM + mi * 16 + hi * 8 + r;
        long col = tileN + ni * 16 + cl;
        if (F32OUT) ((float*)Cv)[row * N + col] = acc[mi][ni][r];
        else        ((bf16*)Cv)[row * N + col]  = f2bf(acc[mi][ni][r]);
      }
}

// Per-row modality-selected GEMM: C = sel ? A@W1 : A@W0 (selection per M row).
// 2x2 blocking with dual accumulators (8 x v8f = 64 VGPRs of accum).
__global__ __launch_bounds__(128) void gemm_select_kernel(
    const bf16* __restrict__ A, const bf16* __restrict__ W0T,
    const bf16* __restrict__ W1T, const int* __restrict__ mod,
    bf16* __restrict__ C, int M, int N, int K) {
  int lane  = threadIdx.x & 31;
  int wave  = threadIdx.x >> 5;
  int tileN = blockIdx.x * 32;
  int tileM = (blockIdx.y * 4 + wave) * 32;

  v8f acc0[2][2], acc1[2][2];
#pragma unroll
  for (int mi = 0; mi < 2; ++mi)
#pragma unroll
    for (int ni = 0; ni < 2; ++ni) { acc0[mi][ni] = (v8f){}; acc1[mi][ni] = (v8f){}; }

  for (int k0 = 0; k0 < K; k0 += 32) {
    v16bf a[2], b0[2], b1[2];
#pragma unroll
    for (int mi = 0; mi < 2; ++mi)
      a[mi] = load_a_frag(A, K, tileM + mi * 16, k0, lane);
#pragma unroll
    for (int ni = 0; ni < 2; ++ni) {
      b0[ni] = load_b_frag(W0T, K, tileN + ni * 16, k0, lane);
      b1[ni] = load_b_frag(W1T, K, tileN + ni * 16, k0, lane);
    }
#pragma unroll
    for (int mi = 0; mi < 2; ++mi)
#pragma unroll
      for (int ni = 0; ni < 2; ++ni) {
        acc0[mi][ni] = wmma_bf16(a[mi], b0[ni], acc0[mi][ni]);
        acc1[mi][ni] = wmma_bf16(a[mi], b1[ni], acc1[mi][ni]);
      }
  }

  int hi = lane >> 4, cl = lane & 15;
#pragma unroll
  for (int mi = 0; mi < 2; ++mi)
#pragma unroll
    for (int ni = 0; ni < 2; ++ni)
#pragma unroll
      for (int r = 0; r < 8; ++r) {
        long row = tileM + mi * 16 + hi * 8 + r;
        long col = tileN + ni * 16 + cl;
        float v = (mod[row] == 1) ? acc1[mi][ni][r] : acc0[mi][ni][r];
        C[row * N + col] = f2bf(v);
      }
}

// ---------------- RoPE + relayout -------------------------------------------
// q_pre:(B*S, H*HD) k_pre/v_pre:(B*S, KV*HD) bf16 -> q_b:(b,h,s,hd),
// k_b:(b,kv,s,hd), vT_b:(b,kv,hd,s), all bf16.
__global__ void rope_relayout_kernel(
    const bf16* __restrict__ q_pre, const bf16* __restrict__ k_pre,
    const bf16* __restrict__ v_pre, const float* __restrict__ cosT,
    const float* __restrict__ sinT, bf16* __restrict__ q_b,
    bf16* __restrict__ k_b, bf16* __restrict__ vT_b) {
  int i = blockIdx.x * blockDim.x + threadIdx.x;   // over B*S*H*HD
  if (i >= B_ * S_ * H_ * HD_) return;
  int d = i % HD_;
  int t = i / HD_;
  int h = t % H_;
  int bs = t / H_;
  int s = bs % S_;
  int b = bs / S_;
  int dro = (d < HD_ / 2) ? d + HD_ / 2 : d - HD_ / 2;
  float c  = cosT[s * HD_ + d];
  float sn = sinT[s * HD_ + d];

  float qx = bf2f(q_pre[(long)bs * NQ_ + h * HD_ + d]);
  float qr = bf2f(q_pre[(long)bs * NQ_ + h * HD_ + dro]);
  float qo = (d < HD_ / 2) ? (qx * c - qr * sn) : (qx * c + qr * sn);
  q_b[(((long)b * H_ + h) * S_ + s) * HD_ + d] = f2bf(qo);

  if (h < KV_) {
    float kx = bf2f(k_pre[(long)bs * NKV_ + h * HD_ + d]);
    float kr = bf2f(k_pre[(long)bs * NKV_ + h * HD_ + dro]);
    float ko = (d < HD_ / 2) ? (kx * c - kr * sn) : (kx * c + kr * sn);
    k_b[(((long)b * KV_ + h) * S_ + s) * HD_ + d] = f2bf(ko);
    // V transposed: (b, kv, hd, s) so P@V B-fragments are contiguous loads
    vT_b[(((long)b * KV_ + h) * HD_ + d) * S_ + s] =
        v_pre[(long)bs * NKV_ + h * HD_ + d];
  }
}

// ---------------- flash attention (causal, GQA) -----------------------------
// grid: (S/64, H, B); block 128 = 4 waves; each wave owns 16 query rows.
__global__ __launch_bounds__(128) void flash_attn_kernel(
    const bf16* __restrict__ q_b, const bf16* __restrict__ k_b,
    const bf16* __restrict__ vT_b, bf16* __restrict__ ctx_b) {
  int lane = threadIdx.x & 31;
  int wave = threadIdx.x >> 5;
  int b = blockIdx.z, h = blockIdx.y;
  int kv = h / (H_ / KV_);
  int qBase = blockIdx.x * 64 + wave * 16;

  const bf16* Q  = q_b  + (((long)b * H_  + h)  * S_)  * HD_;
  const bf16* Kt = k_b  + (((long)b * KV_ + kv) * S_)  * HD_;
  const bf16* Vt = vT_b + (((long)b * KV_ + kv) * HD_) * S_;

  __shared__ __align__(32) bf16 pLds[4][16][32];
  bf16 (*P)[32] = pLds[wave];

  // Q fragments for the 4 K=32 steps covering HD=128 (resident for whole loop)
  v16bf qf[4];
#pragma unroll
  for (int c = 0; c < 4; ++c) qf[c] = load_a_frag(Q, HD_, qBase, c * 32, lane);

  v8f zero = {};
  v8f ctxAcc[8];
#pragma unroll
  for (int t = 0; t < 8; ++t) ctxAcc[t] = zero;

  float m[8], l[8];
#pragma unroll
  for (int r = 0; r < 8; ++r) { m[r] = -1.0e30f; l[r] = 0.0f; }

  int hi = lane >> 4;
  int colLane = lane & 15;
  const float scale = 0.08838834764831845f;   // 1/sqrt(128)
  int kEnd = qBase + 16;                      // causal bound (exclusive)

  for (int j = 0; j < kEnd; j += 32) {
    // ---- scores: two 16x16 tiles over key columns j..j+31 ----
    v8f s0 = zero, s1 = zero;
#pragma unroll
    for (int c = 0; c < 4; ++c) {
      v16bf b0 = load_b_frag(Kt, HD_, j,      c * 32, lane);
      v16bf b1 = load_b_frag(Kt, HD_, j + 16, c * 32, lane);
      s0 = wmma_bf16(qf[c], b0, s0);
      s1 = wmma_bf16(qf[c], b1, s1);
    }
    // ---- online softmax (16-lane groups hold 8 rows each) ----
#pragma unroll
    for (int r = 0; r < 8; ++r) {
      int row = qBase + hi * 8 + r;
      int c0 = j + colLane, c1 = j + 16 + colLane;
      float v0 = (c0 <= row) ? s0[r] * scale : -1.0e30f;
      float v1 = (c1 <= row) ? s1[r] * scale : -1.0e30f;
      float rm = fmaxf(v0, v1);
#pragma unroll
      for (int o = 1; o < 16; o <<= 1) rm = fmaxf(rm, __shfl_xor(rm, o, 16));
      float mNew  = fmaxf(m[r], rm);
      float alpha = __expf(m[r] - mNew);
      float e0 = __expf(v0 - mNew);
      float e1 = __expf(v1 - mNew);
      float rs = e0 + e1;
#pragma unroll
      for (int o = 1; o < 16; o <<= 1) rs += __shfl_xor(rs, o, 16);
      l[r] = l[r] * alpha + rs;
      m[r] = mNew;
#pragma unroll
      for (int t = 0; t < 8; ++t) ctxAcc[t][r] *= alpha;
      P[hi * 8 + r][colLane]      = f2bf(e0);
      P[hi * 8 + r][16 + colLane] = f2bf(e1);
    }
    __builtin_amdgcn_wave_barrier();   // keep DS store->load ordering (in-order per wave)

    // ---- ctx += P(16x32) @ V(32xHD): 8 tiles over hd ----
    v16bf pf = load_a_frag(&P[0][0], 32, 0, 0, lane);
#pragma unroll
    for (int t = 0; t < 8; ++t) {
      v16bf vf = load_b_frag(Vt, S_, t * 16, j, lane);
      ctxAcc[t] = wmma_bf16(pf, vf, ctxAcc[t]);
    }
  }

  // ---- epilogue: normalize and store ctx row-major (B*S, H*HD) bf16 ----
#pragma unroll
  for (int r = 0; r < 8; ++r) {
    int srow = qBase + hi * 8 + r;
    float inv = 1.0f / l[r];
#pragma unroll
    for (int t = 0; t < 8; ++t) {
      ctx_b[((long)(b * S_ + srow)) * NQ_ + h * HD_ + t * 16 + colLane] =
          f2bf(ctxAcc[t][r] * inv);
    }
  }
}

// ---------------------------------------------------------------------------
extern "C" void kernel_launch(void* const* d_in, const int* in_sizes, int n_in,
                              void* d_out, int out_size, void* d_ws, size_t ws_size,
                              hipStream_t stream) {
  (void)in_sizes; (void)n_in; (void)out_size; (void)ws_size;
  const float* x   = (const float*)d_in[0];
  const int*   mod = (const int*)d_in[1];
  const int*   pos = (const int*)d_in[2];
  const float* Wq  = (const float*)d_in[3];
  const float* Wk0 = (const float*)d_in[4];
  const float* Wk1 = (const float*)d_in[5];
  const float* Wv0 = (const float*)d_in[6];
  const float* Wv1 = (const float*)d_in[7];
  const float* Wo  = (const float*)d_in[8];
  float* out = (float*)d_out;

  char* w = (char*)d_ws;
  size_t off = 0;
  auto take = [&](size_t bytes) -> char* {
    char* p = w + off;
    off += (bytes + 255) & ~(size_t)255;
    return p;
  };

  bf16* xb    = (bf16*)take((size_t)M_ * D_ * 2);
  bf16* WqT   = (bf16*)take((size_t)NQ_ * D_ * 2);
  bf16* Wk0T  = (bf16*)take((size_t)NKV_ * D_ * 2);
  bf16* Wk1T  = (bf16*)take((size_t)NKV_ * D_ * 2);
  bf16* Wv0T  = (bf16*)take((size_t)NKV_ * D_ * 2);
  bf16* Wv1T  = (bf16*)take((size_t)NKV_ * D_ * 2);
  bf16* WoT   = (bf16*)take((size_t)D_ * NQ_ * 2);
  float* cosT = (float*)take((size_t)S_ * HD_ * 4);
  float* sinT = (float*)take((size_t)S_ * HD_ * 4);
  bf16* q_pre = (bf16*)take((size_t)M_ * NQ_ * 2);
  bf16* k_pre = (bf16*)take((size_t)M_ * NKV_ * 2);
  bf16* v_pre = (bf16*)take((size_t)M_ * NKV_ * 2);
  bf16* q_b   = (bf16*)take((size_t)M_ * NQ_ * 2);
  bf16* k_b   = (bf16*)take((size_t)M_ * NKV_ * 2);
  bf16* vT_b  = (bf16*)take((size_t)M_ * NKV_ * 2);
  bf16* ctx_b = (bf16*)take((size_t)M_ * NQ_ * 2);

  const int TPB = 256;
  auto blocks = [&](long n) { return (unsigned)((n + TPB - 1) / TPB); };

  // prep: bf16 conversions + K-major weights + rope tables
  convert_f32_bf16_kernel<<<blocks((long)M_ * D_), TPB, 0, stream>>>(x, xb, M_ * D_);
  transpose_to_bf16_kernel<<<blocks((long)D_ * NQ_),  TPB, 0, stream>>>(Wq,  WqT,  D_, NQ_);
  transpose_to_bf16_kernel<<<blocks((long)D_ * NKV_), TPB, 0, stream>>>(Wk0, Wk0T, D_, NKV_);
  transpose_to_bf16_kernel<<<blocks((long)D_ * NKV_), TPB, 0, stream>>>(Wk1, Wk1T, D_, NKV_);
  transpose_to_bf16_kernel<<<blocks((long)D_ * NKV_), TPB, 0, stream>>>(Wv0, Wv0T, D_, NKV_);
  transpose_to_bf16_kernel<<<blocks((long)D_ * NKV_), TPB, 0, stream>>>(Wv1, Wv1T, D_, NKV_);
  transpose_to_bf16_kernel<<<blocks((long)NQ_ * D_),  TPB, 0, stream>>>(Wo,  WoT,  NQ_, D_);
  rope_tables_kernel<<<blocks((long)S_ * HD_), TPB, 0, stream>>>(pos, cosT, sinT);

  // projections (register-blocked WMMA)
  // Q: wave tile 64x32, block tile 256x32 -> grid (4096/32, 4096/256)
  gemm_tiled_kernel<4, 2, false><<<dim3(NQ_ / 32, M_ / 256), 128, 0, stream>>>(
      xb, WqT, (void*)q_pre, M_, NQ_, D_);
  // K/V with per-row modality select: wave tile 32x32, block tile 128x32
  gemm_select_kernel<<<dim3(NKV_ / 32, M_ / 128), 128, 0, stream>>>(
      xb, Wk0T, Wk1T, mod, k_pre, M_, NKV_, D_);
  gemm_select_kernel<<<dim3(NKV_ / 32, M_ / 128), 128, 0, stream>>>(
      xb, Wv0T, Wv1T, mod, v_pre, M_, NKV_, D_);

  // RoPE + layouts
  rope_relayout_kernel<<<blocks((long)B_ * S_ * H_ * HD_), TPB, 0, stream>>>(
      q_pre, k_pre, v_pre, cosT, sinT, q_b, k_b, vT_b);

  // flash attention (WMMA QK^T and PV)
  flash_attn_kernel<<<dim3(S_ / 64, H_, B_), 128, 0, stream>>>(q_b, k_b, vT_b, ctx_b);

  // output projection (register-blocked WMMA, fp32 out)
  gemm_tiled_kernel<4, 2, true><<<dim3(D_ / 32, M_ / 256), 128, 0, stream>>>(
      ctx_b, WoT, (void*)out, M_, D_, NQ_);
}
